// MiniBatchEGNNTrain_35021163331775
// MI455X (gfx1250) — matrix operand order repaired and checked by reading, not compiled, not run on hardware
//
#include <hip/hip_runtime.h>
#include <stdint.h>

typedef __attribute__((ext_vector_type(16))) _Float16 v16h;
typedef __attribute__((ext_vector_type(8)))  float    v8f;
typedef __attribute__((ext_vector_type(4)))  float    v4f;

#define HDIM 64
#define KF   4
#define KDIM 256   // KF * D_IN
#define CDIM 40

// Native CDNA5 f32 global atomic add (no return -> STOREcnt; S_ENDPGM waits idle).
// HIP's atomicAdd(float*) may lower to a CAS loop; with 32M scatter-adds per layer
// the native GLOBAL_ATOMIC_ADD_F32 L2 path is mandatory.
__device__ __forceinline__ void atomic_add_f32_native(float* p, float v) {
  uint64_t a = (uint64_t)p;
  asm volatile("global_atomic_add_f32 %0, %1, off" : : "v"(a), "v"(v) : "memory");
}

// ---------------- utility kernels ----------------
__global__ void k_zero(float* __restrict__ p, int n) {
  int i = blockIdx.x * blockDim.x + threadIdx.x;
  int stride = gridDim.x * blockDim.x;
  for (; i < n; i += stride) p[i] = 0.f;
}

__global__ void k_degree(const int* __restrict__ dst, float* __restrict__ cnt, int e) {
  int i = blockIdx.x * blockDim.x + threadIdx.x;
  if (i < e) atomic_add_f32_native(&cnt[dst[i]], 1.0f);
}

// h = relu(s / max(cnt,1)), in place
__global__ void k_norm_relu(float* __restrict__ s, const float* __restrict__ cnt, int total) {
  int i = blockIdx.x * blockDim.x + threadIdx.x;
  if (i < total) {
    float c = cnt[i >> 6];
    c = c < 1.f ? 1.f : c;
    float v = s[i] / c;
    s[i] = v > 0.f ? v : 0.f;
  }
}

// ---------------- edge message GEMM + scatter ----------------
// One wave32 per 16-edge tile: m[16,64] = relu( (e ⊗ h_src)[16,256] @ W[256,64] + b )
// then native-atomic scatter-add into sout[dst].
__global__ __launch_bounds__(256)
void k_layer(const float* __restrict__ hin,   // [N,64]
             const float* __restrict__ ef,    // [E,4]
             const int*   __restrict__ src,
             const int*   __restrict__ dst,
             const float* __restrict__ W,     // [256,64] row-major (row = k*64+d)
             const float* __restrict__ bias,  // [64]
             float*       __restrict__ sout,  // [N,64], pre-zeroed
             int numTiles)
{
  // W pre-fragmentized into per-lane B-operand layout:
  // Wfrag[((c*4+nb)*32 + lane)*16 + j] = W[c*32 + 8*(lane>>4) + (j&7) + 16*(j>>3)][nb*16 + (lane&15)]
  __shared__ _Float16 Wfrag[KDIM * HDIM];  // 32 KB
  for (int i = threadIdx.x; i < KDIM * HDIM; i += blockDim.x) {
    int j    = i & 15;
    int ln   = (i >> 4) & 31;
    int nb   = (i >> 9) & 3;
    int c    = i >> 11;
    int offB = (ln >> 4) << 3;                // 0 or 8 (K sub-offset per lane group)
    int n    = (nb << 4) | (ln & 15);
    int krow = (c << 5) + offB + (j & 7) + ((j >> 3) << 4);
    Wfrag[i] = (_Float16)W[krow * HDIM + n];
  }
  __syncthreads();

  const int lane = threadIdx.x & 31;
  const int wave = threadIdx.x >> 5;
  const int tile = blockIdx.x * (blockDim.x >> 5) + wave;
  if (tile >= numTiles) return;                // wave-uniform: EXEC stays all-1 for WMMA

  const int m     = lane & 15;
  const int msel  = lane >> 4;
  const int off   = msel << 3;                 // A-operand K sub-offset: 0 or 8
  const int ebase = tile << 4;

  const int srow = src[ebase + m];
  const float* hrow = hin + (size_t)srow * HDIM;

  v4f e4 = *(const v4f*)(ef + (size_t)(ebase + m) * KF);

  // Each lane touches only 32 distinct floats of its h_src row:
  //   d = q*16 + off + j, q = 0..3, j = 0..7   (off = 0 or 8 by lane half)
  float xs[32];
#pragma unroll
  for (int q = 0; q < 4; ++q) {
    *(v4f*)&xs[q * 8]     = *(const v4f*)(hrow + q * 16 + off);
    *(v4f*)&xs[q * 8 + 4] = *(const v4f*)(hrow + q * 16 + off + 4);
  }
  // Convert once to f16: xv[0] covers d-halves {off.., 16+off..} (even chunks),
  //                      xv[1] covers {32+off.., 48+off..} (odd chunks).
  v16h xv[2];
#pragma unroll
  for (int p = 0; p < 2; ++p)
#pragma unroll
    for (int j = 0; j < 16; ++j) xv[p][j] = (_Float16)xs[p * 16 + j];

  const _Float16 eh[4] = { (_Float16)e4.x, (_Float16)e4.y,
                           (_Float16)e4.z, (_Float16)e4.w };

  // A fragments: af[c] = e[k=c>>1] * h[d-base=(c&1)*32 ...]  -> 64 v_pk_mul_f16 total
  v16h af[8];
#pragma unroll
  for (int c = 0; c < 8; ++c) af[c] = xv[c & 1] * eh[c >> 1];

  int drows[8];
#pragma unroll
  for (int r = 0; r < 8; ++r) drows[r] = dst[ebase + (msel << 3) + r];

  const int n0 = lane & 15;
#pragma unroll
  for (int nb = 0; nb < 4; ++nb) {
    const float bn = bias[(nb << 4) | n0];
    v8f acc = { bn, bn, bn, bn, bn, bn, bn, bn };
#pragma unroll
    for (int c = 0; c < 8; ++c) {
      const v16h bf = *(const v16h*)&Wfrag[((((c << 2) | nb) << 5) | lane) << 4];
      acc = __builtin_amdgcn_wmma_f32_16x16x32_f16(false, af[c], false, bf,
                                                   (short)0, acc, false, false);
    }
    // relu + scatter-add: C/D layout -> element r: M = r + 8*msel, N = nb*16 + (lane&15)
#pragma unroll
    for (int r = 0; r < 8; ++r) {
      float v = acc[r];
      v = v > 0.f ? v : 0.f;
      atomic_add_f32_native(sout + (size_t)drows[r] * HDIM + ((nb << 4) | n0), v);
    }
  }
}

// ---------------- final FC: out[N,40] = h[N,64] @ Wfc[64,40] + bfc ----------------
__global__ __launch_bounds__(256)
void k_fc(const float* __restrict__ hin,  // [N,64]
          const float* __restrict__ Wfc,  // [64,40]
          const float* __restrict__ bfc,  // [40]
          float*       __restrict__ out,  // [N,40]
          int numTiles)
{
  __shared__ _Float16 Wfrag[2 * 3 * 32 * 16];  // K-chunks=2, N-blocks=3 (40 padded to 48)
  for (int i = threadIdx.x; i < 2 * 3 * 32 * 16; i += blockDim.x) {
    int j    = i & 15;
    int ln   = (i >> 4) & 31;
    int rest = i >> 9;
    int nb   = rest % 3;
    int c    = rest / 3;
    int offB = (ln >> 4) << 3;
    int n    = (nb << 4) | (ln & 15);
    int krow = (c << 5) + offB + (j & 7) + ((j >> 3) << 4);
    Wfrag[i] = (n < CDIM) ? (_Float16)Wfc[krow * CDIM + n] : (_Float16)0.f;
  }
  __syncthreads();

  const int lane = threadIdx.x & 31;
  const int wave = threadIdx.x >> 5;
  const int tile = blockIdx.x * (blockDim.x >> 5) + wave;
  if (tile >= numTiles) return;

  const int m       = lane & 15;
  const int msel    = lane >> 4;
  const int off     = msel << 3;
  const int rowbase = tile << 4;
  const float* hrow = hin + (size_t)(rowbase + m) * HDIM;

  float xs[32];
#pragma unroll
  for (int q = 0; q < 4; ++q) {
    *(v4f*)&xs[q * 8]     = *(const v4f*)(hrow + q * 16 + off);
    *(v4f*)&xs[q * 8 + 4] = *(const v4f*)(hrow + q * 16 + off + 4);
  }
  v16h af[2];
#pragma unroll
  for (int c = 0; c < 2; ++c)
#pragma unroll
    for (int j = 0; j < 16; ++j) af[c][j] = (_Float16)xs[c * 16 + j];

  const int n0 = lane & 15;
#pragma unroll
  for (int nb = 0; nb < 3; ++nb) {
    const int n = (nb << 4) | n0;
    const float bn = (n < CDIM) ? bfc[n] : 0.f;
    v8f acc = { bn, bn, bn, bn, bn, bn, bn, bn };
#pragma unroll
    for (int c = 0; c < 2; ++c) {
      const v16h bf = *(const v16h*)&Wfrag[(((c * 3 + nb) << 5) | lane) << 4];
      acc = __builtin_amdgcn_wmma_f32_16x16x32_f16(false, af[c], false, bf,
                                                   (short)0, acc, false, false);
    }
    if (n < CDIM) {
#pragma unroll
      for (int r = 0; r < 8; ++r)
        out[(size_t)(rowbase + (msel << 3) + r) * CDIM + n] = acc[r];
    }
  }
}

// ---------------- launch ----------------
extern "C" void kernel_launch(void* const* d_in, const int* in_sizes, int n_in,
                              void* d_out, int out_size, void* d_ws, size_t ws_size,
                              hipStream_t stream) {
  const float* nodef = (const float*)d_in[0];
  const float* ef    = (const float*)d_in[1];   // [L,E,4]
  const int*   src   = (const int*)d_in[2];
  const int*   dst   = (const int*)d_in[3];
  const float* W0    = (const float*)d_in[4];
  const float* b0    = (const float*)d_in[5];
  const float* W1    = (const float*)d_in[6];
  const float* b1    = (const float*)d_in[7];
  const float* Wfc   = (const float*)d_in[8];
  const float* bfc   = (const float*)d_in[9];
  float* out = (float*)d_out;

  const int E = in_sizes[2];
  const int N = in_sizes[0] / HDIM;

  float* cnt = (float*)d_ws;
  float* sA  = cnt + N;
  float* sB  = sA + (size_t)N * HDIM;

  const int zeroTotal = N + 2 * N * HDIM;
  k_zero<<<1024, 256, 0, stream>>>(cnt, zeroTotal);
  k_degree<<<(E + 255) / 256, 256, 0, stream>>>(dst, cnt, E);

  const int eTiles = E / 16;                   // E = 500000 = 31250 * 16
  const int layerBlocks = (eTiles + 7) / 8;    // 8 waves (tiles) per block

  // Layer 0: h = node_features -> sA
  k_layer<<<layerBlocks, 256, 0, stream>>>(nodef, ef, src, dst, W0, b0, sA, eTiles);
  k_norm_relu<<<(N * HDIM + 255) / 256, 256, 0, stream>>>(sA, cnt, N * HDIM);

  // Layer 1: h = sA -> sB
  k_layer<<<layerBlocks, 256, 0, stream>>>(sA, ef + (size_t)E * KF, src, dst, W1, b1, sB, eTiles);
  k_norm_relu<<<(N * HDIM + 255) / 256, 256, 0, stream>>>(sB, cnt, N * HDIM);

  // Final FC
  const int nTiles = N / 16;                   // N = 50000 = 3125 * 16
  k_fc<<<(nTiles + 7) / 8, 256, 0, stream>>>(sB, Wfc, bfc, out, nTiles);
}
